// NicheTrans_img_58703613002247
// MI455X (gfx1250) — matrix operand order repaired and checked by reading, not compiled, not run on hardware
//
#include <hip/hip_runtime.h>

// NicheTrans heads: out[b,t] = sum_k leaky(BN(f.W1[t,k]+b1)) * W2[t,k] + b2[t]
// Fused bf16-WMMA GEMM (M=8192, N=137*128, K=384) with in-register epilogue.

namespace {
constexpr int BROWS = 8192;
constexpr int TH    = 137;
constexpr int DF    = 384;
constexpr int KH    = 128;
constexpr int KSTEP = 32;            // K-depth of one v_wmma_f32_16x16x32_bf16
constexpr int NKC   = DF / KSTEP;    // 12 k-chunks
constexpr int LSTR  = 40;            // bf16 stride per hidden-unit row in LDS (32 + pad, 80B keeps 16B align)
constexpr float EPSV  = 1e-5f;
constexpr float SLOPE = 0.01f;
}

typedef __attribute__((ext_vector_type(16))) __bf16 v16bf;
typedef __attribute__((ext_vector_type(8)))  __bf16 v8bf;
typedef __attribute__((ext_vector_type(8)))  float  v8f;

__device__ __forceinline__ v8bf cvt8(float4 a, float4 b) {
  v8bf r;
  r[0] = (__bf16)a.x; r[1] = (__bf16)a.y; r[2] = (__bf16)a.z; r[3] = (__bf16)a.w;
  r[4] = (__bf16)b.x; r[5] = (__bf16)b.y; r[6] = (__bf16)b.z; r[7] = (__bf16)b.w;
  return r;
}

__global__ __launch_bounds__(256)
void niche_heads_wmma_kernel(const float* __restrict__ f,
                             const float* __restrict__ W1,
                             const float* __restrict__ b1,
                             const float* __restrict__ gamma,
                             const float* __restrict__ beta,
                             const float* __restrict__ rmean,
                             const float* __restrict__ rvar,
                             const float* __restrict__ W2,
                             const float* __restrict__ b2,
                             float* __restrict__ out)
{
  __shared__ __bf16 Bb[2][KH * LSTR];     // double-buffered W1_t chunk, bf16
  __shared__ float scale_s[KH];
  __shared__ float shift_s[KH];
  __shared__ float w2_s[KH];

  const int t    = blockIdx.x;            // head index (consecutive blocks share f rows in L2)
  const int mblk = blockIdx.y;            // 128-row block of f
  const int tid  = threadIdx.x;
  const int wave = tid >> 5;              // wave32
  const int lane = tid & 31;
  const int l15  = lane & 15;
  const int lhi  = lane >> 4;             // 0 or 1

  const int rowBase = mblk * 128 + wave * 16;

  // --- fold BN(eval) + b1 into per-hidden-unit scale/shift; cache W2 ---
  if (tid < KH) {
    const size_t idx = (size_t)t * KH + tid;
    const float sc = gamma[idx] * rsqrtf(rvar[idx] + EPSV);
    scale_s[tid] = sc;
    shift_s[tid] = (b1[idx] - rmean[idx]) * sc + beta[idx];
    w2_s[tid]    = W2[idx];
  }

  // cooperative W1 chunk loader mapping: thread -> (hidden unit n, 16-wide k-half)
  const int cn = tid >> 1;                // 0..127
  const int ch = tid & 1;                 // k-half
  const float* w1p  = W1 + ((size_t)t * KH + cn) * DF + ch * 16;
  const int ldsoff  = cn * LSTR + ch * 16;

  // A rows: 16-bit A 16x32 layout -> lane(0..15)=row, K 0..7 & 16..23;
  // lanes 16..31 same rows, K 8..15 & 24..31. Fold the +8 into the row pointer.
  const float* arow = f + (size_t)(rowBase + l15) * DF + lhi * 8;

  v8f acc[8] = {};

  float4 pf0, pf1, pf2, pf3;
  {   // prefetch + commit chunk 0 into buffer 0
    const float4* p = (const float4*)(w1p);
    pf0 = p[0]; pf1 = p[1]; pf2 = p[2]; pf3 = p[3];
    *(v8bf*)&Bb[0][ldsoff]     = cvt8(pf0, pf1);
    *(v8bf*)&Bb[0][ldsoff + 8] = cvt8(pf2, pf3);
  }
  __syncthreads();

  for (int kc = 0; kc < NKC; ++kc) {
    const int cur = kc & 1;

    // prefetch next W1 chunk from global while this chunk computes
    if (kc + 1 < NKC) {
      const float4* p = (const float4*)(w1p + (kc + 1) * KSTEP);
      pf0 = p[0]; pf1 = p[1]; pf2 = p[2]; pf3 = p[3];
    }

    // A fragment straight from global (rows are wave-private; hits L2 across heads)
    v16bf aF;
    {
      const float4* q  = (const float4*)(arow + kc * KSTEP);
      const float4* q2 = (const float4*)(arow + kc * KSTEP + 16);
      aF = __builtin_shufflevector(cvt8(q[0], q[1]), cvt8(q2[0], q2[1]),
                                   0,1,2,3,4,5,6,7,8,9,10,11,12,13,14,15);
    }

    #pragma unroll
    for (int j = 0; j < 8; ++j) {
      // 16-bit B 32x16 layout: lane = column, lanes 0..15 hold K 0..7 & 16..23, 16..31 hold K 8..15 & 24..31
      const __bf16* src = &Bb[cur][(j * 16 + l15) * LSTR + lhi * 8];
      v8bf blo = *(const v8bf*)src;
      v8bf bhi = *(const v8bf*)(src + 16);
      v16bf bF = __builtin_shufflevector(blo, bhi,
                                         0,1,2,3,4,5,6,7,8,9,10,11,12,13,14,15);
      acc[j] = __builtin_amdgcn_wmma_f32_16x16x32_bf16(
                   false, aF, false, bF, (short)0, acc[j], false, false);
    }

    if (kc + 1 < NKC) {   // commit prefetched chunk into the other buffer
      *(v8bf*)&Bb[cur ^ 1][ldsoff]     = cvt8(pf0, pf1);
      *(v8bf*)&Bb[cur ^ 1][ldsoff + 8] = cvt8(pf2, pf3);
    }
    __syncthreads();
  }

  // --- fused epilogue: BN + LeakyReLU + dot with W2, reduce over 128 hidden units ---
  // C/D layout: VGPR r, lanes 0..15 -> row r, N=lane; lanes 16..31 -> row 8+r, N=lane-16
  float partial[8] = {0.f, 0.f, 0.f, 0.f, 0.f, 0.f, 0.f, 0.f};
  #pragma unroll
  for (int j = 0; j < 8; ++j) {
    const int n = j * 16 + l15;
    const float sc = scale_s[n];
    const float sh = shift_s[n];
    const float w  = w2_s[n];
    #pragma unroll
    for (int r = 0; r < 8; ++r) {
      float v = acc[j][r] * sc + sh;
      v = (v >= 0.f) ? v : SLOPE * v;
      partial[r] += v * w;
    }
  }
  // butterfly reduce across the 16 lanes of each half-group (masks < 16 stay in-group)
  #pragma unroll
  for (int m = 1; m < 16; m <<= 1) {
    #pragma unroll
    for (int r = 0; r < 8; ++r)
      partial[r] += __shfl_xor(partial[r], m, 32);
  }
  if (l15 == 0) {
    const int rb = rowBase + lhi * 8;
    const float bb = b2[t];
    #pragma unroll
    for (int r = 0; r < 8; ++r)
      out[(size_t)(rb + r) * TH + t] = partial[r] + bb;
  }
}

extern "C" void kernel_launch(void* const* d_in, const int* in_sizes, int n_in,
                              void* d_out, int out_size, void* d_ws, size_t ws_size,
                              hipStream_t stream) {
  (void)in_sizes; (void)n_in; (void)out_size; (void)d_ws; (void)ws_size;
  const float* f     = (const float*)d_in[0];
  const float* W1    = (const float*)d_in[1];
  const float* b1    = (const float*)d_in[2];
  const float* gamma = (const float*)d_in[3];
  const float* beta  = (const float*)d_in[4];
  const float* rmean = (const float*)d_in[5];
  const float* rvar  = (const float*)d_in[6];
  const float* W2    = (const float*)d_in[7];
  const float* b2    = (const float*)d_in[8];
  float* out = (float*)d_out;

  dim3 grid(TH, BROWS / 128);   // 137 x 64 workgroups
  dim3 block(256);              // 8 wave32s
  niche_heads_wmma_kernel<<<grid, block, 0, stream>>>(
      f, W1, b1, gamma, beta, rmean, rvar, W2, b2, out);
}
